// GNN_51539607552499
// MI455X (gfx1250) — matrix operand order repaired and checked by reading, not compiled, not run on hardware
//
#include <hip/hip_runtime.h>

// ============================================================================
// CDNA5 / gfx1250 implementation of the part-graph inference network.
// Core compute: implicit-im2col / implicit-deformable-gather GEMMs on
// v_wmma_f32_16x16x32_f16 (f16 staged operands, f32 accumulation).
// Round 2: branch-free predicated gathers (batched loads, no exec-branch
// serialization), incremental k->(c,t9) decomposition (no integer division),
// pre-converted f16 weight arena with direct-from-global A fragments (no A
// tile in LDS), global_prefetch on the streaming 1x1 path.
// ============================================================================

namespace {

constexpr int HWSZ = 9216;        // 96*96
constexpr int NT   = HWSZ / 128;  // 72 N-tiles per GEMM
constexpr int CAPH = 32 * 2664 + 32;  // f16 weight arena (max Ktot=296*9)

typedef _Float16 f16;
typedef __attribute__((ext_vector_type(16))) _Float16 v16h;
typedef __attribute__((ext_vector_type(8)))  float    v8f;

__device__ __forceinline__ float sigm(float v) { return 1.0f / (1.0f + __expf(-v)); }

// ---------------------------------------------------------------------------
// OUT[M, HWSZ] = Wt16[M, Ktot] x B(X)[Ktot, HWSZ] (+bias) (+epilogue)
//   MODE 0: 3x3 zero-pad im2col of X[Cin,96,96]; Ktot=Cin*9, k=c*9+t9
//   MODE 1: deformable bilinear sample * mask; offm[27,HW]
//   MODE 2: 1x1 conv; Ktot=Cin
// Wt16: f16 weights, zero-padded so rows m<32 and k<kk+32 are always readable.
// grid = 72, block = 256 (8 waves: 2 M-rows x 4 N-cols of 16x32 tiles).
// ---------------------------------------------------------------------------
template <int MODE>
__global__ void __launch_bounds__(256)
k_conv_gemm(const float* __restrict__ X, const f16* __restrict__ Wt,
            const float* __restrict__ bias, const float* __restrict__ offm,
            float* __restrict__ out, int M, int Cin, int epi)
{
  const int Ktot = (MODE == 2) ? Cin : Cin * 9;
  __shared__ f16 Bl[128][40];   // B tile [n][k], 80B row stride (16B aligned)

  const int tid  = threadIdx.x;
  const int lane = tid & 31, wid = tid >> 5;
  const int n0   = blockIdx.x * 128;
  const int mrow = (wid >> 2) * 16;    // 0 / 16
  const int ncol = (wid & 3) * 32;     // 0,32,64,96

  // --- B-fill: thread-fixed pixel, incremental k=(c,t9), stride +2 ---
  const int nl = tid & 127;
  const int hi = tid >> 7;             // 0/1
  const int n  = n0 + nl;
  const int y  = n / 96, x = n - y * 96;
  int kcur = hi, cch = 0, t9 = hi;

  // --- A fragment: direct dword loads from f16 weight arena ---
  const unsigned* Wd = (const unsigned*)Wt;
  const int am    = mrow + (lane & 15);
  const int akb2  = (lane < 16) ? 0 : 8;          // dword offset (half offset 0/16)
  const int abase = (am * Ktot) >> 1;             // Ktot always even

  v8f c0 = {}, c1 = {};

  for (int kk = 0; kk < Ktot; kk += 32) {
    // ---------------- stage B tile (branch-free) ----------------
#pragma unroll 4
    for (int i = 0; i < 16; ++i) {
      const int k = kcur;
      const float valid = (k < Ktot) ? 1.0f : 0.0f;
      float v;
      if (MODE == 2) {
        int kc = (k < Ktot) ? k : (Ktot - 1);
        v = X[kc * HWSZ + n] * valid;
        if (k + 32 < Ktot) __builtin_prefetch(&X[(k + 32) * HWSZ + n], 0, 3);
      } else {
        const int cc = (cch < Cin) ? cch : (Cin - 1);
        const int dy = ((0x2A540 >> (2 * t9)) & 3) - 1;   // t9/3 - 1
        const int dx = ((0x24924 >> (2 * t9)) & 3) - 1;   // t9%3 - 1
        if (MODE == 0) {
          int yy = y + dy, xx = x + dx;
          float inb = ((unsigned)yy < 96u && (unsigned)xx < 96u) ? valid : 0.0f;
          int yc = min(max(yy, 0), 95), xc = min(max(xx, 0), 95);
          v = X[cc * HWSZ + yc * 96 + xc] * inb;
        } else {
          const float* op = offm + n;
          float py = op[(2 * t9) * HWSZ] + (float)(y + dy);
          float px = op[(2 * t9 + 1) * HWSZ] + (float)(x + dx);
          float mk = op[(18 + t9) * HWSZ];
          float y0f = floorf(py), x0f = floorf(px);
          int y0 = (int)y0f, x0 = (int)x0f;
          float wy = py - y0f, wx = px - x0f;
          float by0 = ((unsigned)y0 < 96u) ? 1.0f : 0.0f;
          float by1 = ((unsigned)(y0 + 1) < 96u) ? 1.0f : 0.0f;
          float bx0 = ((unsigned)x0 < 96u) ? 1.0f : 0.0f;
          float bx1 = ((unsigned)(x0 + 1) < 96u) ? 1.0f : 0.0f;
          int y0c = min(max(y0, 0), 95), y1c = min(max(y0 + 1, 0), 95);
          int x0c = min(max(x0, 0), 95), x1c = min(max(x0 + 1, 0), 95);
          const float* Xc = X + cc * HWSZ;
          float v00 = Xc[y0c * 96 + x0c] * (by0 * bx0);
          float v01 = Xc[y0c * 96 + x1c] * (by0 * bx1);
          float v10 = Xc[y1c * 96 + x0c] * (by1 * bx0);
          float v11 = Xc[y1c * 96 + x1c] * (by1 * bx1);
          v = ((v00 * (1.0f - wx) + v01 * wx) * (1.0f - wy) +
               (v10 * (1.0f - wx) + v11 * wx) * wy) * (mk * valid);
        }
      }
      Bl[nl][k - kk] = (f16)v;
      kcur += 2;
      t9 += 2;
      if (t9 >= 9) { t9 -= 9; cch += 1; }   // branchless selects after codegen
    }
    __syncthreads();

    // ---------------- A fragment: direct from global (dwords) ----------------
    union AU { unsigned u[8]; v16h v; } au;
    const int adw = abase + (kk >> 1) + akb2;
#pragma unroll
    for (int j = 0; j < 4; ++j) {
      au.u[j]     = Wd[adw + j];          // halves akb+2j, akb+2j+1
      au.u[4 + j] = Wd[adw + 4 + j];      // halves 8+akb+2j, +1
    }
    v16h a = au.v;

    // ---------------- B fragments from LDS ----------------
    const int bn_ = lane & 15;
    const int bkb = (lane < 16) ? 0 : 16;
    v16h b0, b1;
#pragma unroll
    for (int j = 0; j < 8; ++j) {
      b0[2 * j]     = Bl[ncol + bn_][bkb + 2 * j];
      b0[2 * j + 1] = Bl[ncol + bn_][bkb + 2 * j + 1];
      b1[2 * j]     = Bl[ncol + 16 + bn_][bkb + 2 * j];
      b1[2 * j + 1] = Bl[ncol + 16 + bn_][bkb + 2 * j + 1];
    }
    c0 = __builtin_amdgcn_wmma_f32_16x16x32_f16(false, a, false, b0, (short)0, c0, false, false);
    c1 = __builtin_amdgcn_wmma_f32_16x16x32_f16(false, a, false, b1, (short)0, c1, false, false);
    __syncthreads();
  }

  // ---- store: C/D layout VGPR v -> M = v + (lane<16 ? 0 : 8), N = lane&15
  const int nb  = n0 + ncol + (lane & 15);
  const int mhi = (lane < 16) ? 0 : 8;
#pragma unroll
  for (int v = 0; v < 8; ++v) {
    int m = mrow + mhi + v;
    if (m < M) {
      float r0 = c0[v], r1 = c1[v];
      if (bias) { float bb = bias[m]; r0 += bb; r1 += bb; }
      if (epi == 1) { r0 = sigm(r0); r1 = sigm(r1); }
      out[m * HWSZ + nb]      = r0;
      out[m * HWSZ + nb + 16] = r1;
    }
  }
}

// ----------------------------- small kernels -------------------------------
__global__ void k_cvt_f16(const float* __restrict__ src, f16* __restrict__ dst,
                          int n, int ntot) {
  int i = blockIdx.x * 256 + threadIdx.x;
  if (i < ntot) dst[i] = (i < n) ? (f16)src[i] : (f16)0.0f;
}

__global__ void k_sigmoid_ip(float* p, int n) {
  int i = blockIdx.x * 256 + threadIdx.x;
  if (i < n) p[i] = sigm(p[i]);
}

__global__ void k_bn_stats(const float* __restrict__ x, float* __restrict__ mean,
                           float* __restrict__ rstd) {
  int c = blockIdx.x;
  __shared__ float s1[256], s2[256];
  float a = 0.f, b = 0.f;
  for (int i = threadIdx.x; i < HWSZ; i += 256) {
    float v = x[c * HWSZ + i]; a += v; b += v * v;
  }
  s1[threadIdx.x] = a; s2[threadIdx.x] = b; __syncthreads();
  for (int s = 128; s > 0; s >>= 1) {
    if (threadIdx.x < s) { s1[threadIdx.x] += s1[threadIdx.x + s]; s2[threadIdx.x] += s2[threadIdx.x + s]; }
    __syncthreads();
  }
  if (threadIdx.x == 0) {
    float m = s1[0] * (1.0f / HWSZ);
    float v = s2[0] * (1.0f / HWSZ) - m * m;
    mean[c] = m; rstd[c] = rsqrtf(v + 1e-5f);
  }
}

__global__ void k_bn_relu(const float* __restrict__ x, const float* __restrict__ mean,
                          const float* __restrict__ rstd, const float* __restrict__ g,
                          const float* __restrict__ b, float* __restrict__ y, int n) {
  int i = blockIdx.x * 256 + threadIdx.x;
  if (i >= n) return;
  int c = i / HWSZ;
  float v = (x[i] - mean[c]) * rstd[c] * g[c] + b[c];
  y[i] = v > 0.f ? v : 0.f;
}

__global__ void k_bcast_mul(const float* __restrict__ a1, const float* __restrict__ x,
                            float* __restrict__ y, int n) {
  int i = blockIdx.x * 256 + threadIdx.x;
  if (i < n) y[i] = a1[i % HWSZ] * x[i];
}

__global__ void k_pdfin(const float* __restrict__ a1, const float* __restrict__ h,
                        float* __restrict__ y, int n) {
  int i = blockIdx.x * 256 + threadIdx.x;
  if (i < n) y[i] = (2.0f - a1[i % HWSZ]) * h[i];
}

__global__ void k_gate(const float* __restrict__ g1, const float* __restrict__ x,
                       const float* __restrict__ o, float* __restrict__ y, int n) {
  int i = blockIdx.x * 256 + threadIdx.x;
  if (i >= n) return;
  float g = g1[i % HWSZ];
  float v = g * x[i] + (1.0f - g) * o[i];
  y[i] = v > 0.f ? v : 0.f;
}

__global__ void k_add_ip(float* __restrict__ y, const float* __restrict__ x, int n) {
  int i = blockIdx.x * 256 + threadIdx.x;
  if (i < n) y[i] += x[i];
}

__global__ void k_avg2(const float* __restrict__ a, const float* __restrict__ b,
                       float* __restrict__ y, int n) {
  int i = blockIdx.x * 256 + threadIdx.x;
  if (i < n) y[i] = 0.5f * (a[i] + b[i]);
}

// ----------------------------- host-side glue ------------------------------
struct DF { const float *woff, *boff, *w, *b; };
struct BN { const float *g, *b; };
struct CU { DF u1; BN bn1; DF u2; BN bn2; DF gate; };
struct PD { DF d1; BN bn1; DF d2; BN bn2; const float *aw, *ab; };

struct Ctx {
  hipStream_t s;
  float *CAT, *OFF, *T20a, *T20b, *T10a, *T10b, *G1, *A1, *MEAN, *RSTD, *MSGT;
  f16*  WF16;
};

inline int cdiv(int a, int b) { return (a + b - 1) / b; }

void rungemm(Ctx& c, int mode, const float* X, const float* W, const float* bias,
             const float* off, float* out, int M, int Cin, int epi) {
  int Ktot = (mode == 2) ? Cin : Cin * 9;
  k_cvt_f16<<<cdiv(CAPH, 256), 256, 0, c.s>>>(W, c.WF16, M * Ktot, CAPH);
  if (mode == 0)
    k_conv_gemm<0><<<NT, 256, 0, c.s>>>(X, c.WF16, bias, off, out, M, Cin, epi);
  else if (mode == 1)
    k_conv_gemm<1><<<NT, 256, 0, c.s>>>(X, c.WF16, bias, off, out, M, Cin, epi);
  else
    k_conv_gemm<2><<<NT, 256, 0, c.s>>>(X, c.WF16, bias, off, out, M, Cin, epi);
}

void dfconv(Ctx& c, const float* X, int Cin, int Cout, const DF& p, float* out, int epi) {
  rungemm(c, 0, X, p.woff, p.boff, nullptr, c.OFF, 27, Cin, 0);
  k_sigmoid_ip<<<cdiv(9 * HWSZ, 256), 256, 0, c.s>>>(c.OFF + 18 * HWSZ, 9 * HWSZ);
  rungemm(c, 1, X, p.w, p.b, c.OFF, out, Cout, Cin, epi);
}

void bnrelu(Ctx& c, const float* x, const BN& bn, int C, float* y) {
  k_bn_stats<<<C, 256, 0, c.s>>>(x, c.MEAN, c.RSTD);
  k_bn_relu<<<cdiv(C * HWSZ, 256), 256, 0, c.s>>>(x, c.MEAN, c.RSTD, bn.g, bn.b, y, C * HWSZ);
}

void composition(Ctx& c, const float* const* srcs, int nparts, const float* w,
                 const BN& bn, float* dst) {
  for (int i = 0; i < nparts; ++i)
    hipMemcpyAsync(c.CAT + i * 10 * HWSZ, srcs[i], 10 * HWSZ * sizeof(float),
                   hipMemcpyDeviceToDevice, c.s);
  rungemm(c, 2, c.CAT, w, nullptr, nullptr, c.T10a, 10, nparts * 10, 0);
  bnrelu(c, c.T10a, bn, 10, dst);
}

void decomposition(Ctx& c, const float* a10, const float* b10, const float* w,
                   const float* bias, float* dst) {
  hipMemcpyAsync(c.CAT, a10, 10 * HWSZ * sizeof(float), hipMemcpyDeviceToDevice, c.s);
  hipMemcpyAsync(c.CAT + 10 * HWSZ, b10, 10 * HWSZ * sizeof(float), hipMemcpyDeviceToDevice, c.s);
  rungemm(c, 2, c.CAT, w, bias, nullptr, dst, 1, 20, 1);
}

void part_dep(Ctx& c, const float* pA, const float* pB, const PD& p, float* dst) {
  hipMemcpyAsync(c.CAT, pA, 10 * HWSZ * sizeof(float), hipMemcpyDeviceToDevice, c.s);
  hipMemcpyAsync(c.CAT + 10 * HWSZ, pB, 10 * HWSZ * sizeof(float), hipMemcpyDeviceToDevice, c.s);
  dfconv(c, c.CAT, 20, 20, p.d1, c.T20a, 0);
  bnrelu(c, c.T20a, p.bn1, 20, c.T20b);
  dfconv(c, c.T20b, 20, 10, p.d2, c.T10a, 0);
  bnrelu(c, c.T10a, p.bn2, 10, c.T10b);
  rungemm(c, 2, pA, p.aw, p.ab, nullptr, c.A1, 1, 10, 1);
  k_pdfin<<<cdiv(10 * HWSZ, 256), 256, 0, c.s>>>(c.A1, c.T10b, dst, 10 * HWSZ);
}

void conv_update(Ctx& c, const float* fea, const float* x, const float* const* msgs,
                 int nm, const CU& p, float* dst) {
  hipMemcpyAsync(c.CAT, fea, 256 * HWSZ * sizeof(float), hipMemcpyDeviceToDevice, c.s);
  hipMemcpyAsync(c.CAT + 256 * HWSZ, x, 10 * HWSZ * sizeof(float), hipMemcpyDeviceToDevice, c.s);
  for (int i = 0; i < nm; ++i)
    hipMemcpyAsync(c.CAT + (266 + 10 * i) * HWSZ, msgs[i], 10 * HWSZ * sizeof(float),
                   hipMemcpyDeviceToDevice, c.s);
  int Cin = 256 + 10 * (1 + nm);
  dfconv(c, c.CAT, Cin, 20, p.u1, c.T20a, 0);
  bnrelu(c, c.T20a, p.bn1, 20, c.T20b);
  dfconv(c, c.T20b, 20, 1, p.gate, c.G1, 1);
  dfconv(c, c.T20b, 20, 10, p.u2, c.T10a, 0);
  bnrelu(c, c.T10a, p.bn2, 10, c.T10b);
  k_gate<<<cdiv(10 * HWSZ, 256), 256, 0, c.s>>>(c.G1, x, c.T10b, dst, 10 * HWSZ);
}

} // namespace

extern "C" void kernel_launch(void* const* d_in, const int* in_sizes, int n_in,
                              void* d_out, int out_size, void* d_ws, size_t ws_size,
                              hipStream_t stream) {
  (void)in_sizes; (void)n_in; (void)out_size; (void)ws_size;

  // ---- inputs 0..5 ----
  const float* xp_fea  = (const float*)d_in[0];
  const float* xh_fea  = (const float*)d_in[1];
  const float* xf_fea  = (const float*)d_in[2];
  const float* xp_list = (const float*)d_in[3];   // (6,1,10,96,96)
  const float* xh_list = (const float*)d_in[4];   // (2,1,10,96,96)
  const float* xf      = (const float*)d_in[5];   // (1,10,96,96)

  // ---- params: leaves in make_params() dict-construction order ----
  int pi = 6;
  auto nx = [&]() { return (const float*)d_in[pi++]; };
  auto rdBN = [&]() { BN b; b.g = nx(); b.b = nx(); return b; };
  auto rdDF = [&](bool bias) { DF d; d.woff = nx(); d.boff = nx(); d.w = nx();
                               d.b = bias ? nx() : nullptr; return d; };
  auto rdCU = [&]() { CU u; u.u1 = rdDF(false); u.bn1 = rdBN(); u.u2 = rdDF(false);
                      u.bn2 = rdBN(); u.gate = rdDF(true); return u; };
  auto rdPD = [&]() { PD p; p.d1 = rdDF(false); p.bn1 = rdBN(); p.d2 = rdDF(false);
                      p.bn2 = rdBN(); p.aw = nx(); p.ab = nx(); return p; };

  const float* comphW = nx(); BN comphBN = rdBN();
  const float* comppW = nx(); BN comppBN = rdBN();
  CU updF = rdCU();
  const float* dfhuW = nx(); const float* dfhuB = nx();
  const float* dfhlW = nx(); const float* dfhlB = nx();
  const float* compuW = nx(); BN compuBN = rdBN();
  const float* complW = nx(); BN complBN = rdBN();
  PD dpU = rdPD();
  PD dpL = rdPD();
  CU updU = rdCU();
  PD dpE[10];  for (int e = 0; e < 10; ++e) dpE[e] = rdPD();
  const float *dhpW[6], *dhpB[6]; for (int i = 0; i < 6; ++i) { dhpW[i] = nx(); dhpB[i] = nx(); }
  const float *dfpW[6], *dfpB[6]; for (int i = 0; i < 6; ++i) { dfpW[i] = nx(); dfpB[i] = nx(); }
  CU updP[6]; for (int i = 0; i < 6; ++i) updP[i] = rdCU();

  // ---- workspace (bump allocator; ~25 MB total) ----
  char* wsp = (char*)d_ws;
  size_t off = 0;
  auto alloc = [&](size_t nfloats) {
    float* r = (float*)(wsp + off);
    off += ((nfloats * sizeof(float) + 255) / 256) * 256;
    return r;
  };
  Ctx c; c.s = stream;
  c.CAT  = alloc(296 * HWSZ);
  c.OFF  = alloc(27 * HWSZ);
  c.T20a = alloc(20 * HWSZ);
  c.T20b = alloc(20 * HWSZ);
  c.T10a = alloc(10 * HWSZ);
  c.T10b = alloc(10 * HWSZ);
  c.G1   = alloc(HWSZ);
  c.A1   = alloc(HWSZ);
  c.MEAN = alloc(32);
  c.RSTD = alloc(32);
  c.MSGT = alloc(10 * HWSZ);
  c.WF16 = (f16*)alloc(CAPH / 2 + 64);
  float* comp_h = alloc(10 * HWSZ);
  float* comp_p = alloc(10 * HWSZ);
  float* comp_u = alloc(10 * HWSZ);
  float* comp_l = alloc(10 * HWSZ);
  float* dpu    = alloc(10 * HWSZ);
  float* dpl    = alloc(10 * HWSZ);
  float* axf_u  = alloc(10 * HWSZ);
  float* axf_l  = alloc(10 * HWSZ);
  float *decfp[6], *dechp[6], *msgs[6], *attfp[6], *atthp[6];
  for (int i = 0; i < 6; ++i) decfp[i] = alloc(10 * HWSZ);
  for (int i = 0; i < 6; ++i) dechp[i] = alloc(10 * HWSZ);
  for (int i = 0; i < 6; ++i) msgs[i]  = alloc(10 * HWSZ);
  for (int i = 0; i < 6; ++i) attfp[i] = alloc(HWSZ);
  for (int i = 0; i < 6; ++i) atthp[i] = alloc(HWSZ);

  // ---- output regions ----
  float* out    = (float*)d_out;
  float* xh_u_o = out + 60 * HWSZ;
  float* xh_l_o = out + 70 * HWSZ;
  float* xf_n_o = out + 80 * HWSZ;
  float* att_o  = out + 90 * HWSZ;   // [att_u, att_l, att_p x6] = 8 channels

  const float* xh0 = xh_list;
  const float* xh1 = xh_list + 10 * HWSZ;
  const float* xp_[6];
  for (int j = 0; j < 6; ++j) xp_[j] = xp_list + j * 10 * HWSZ;

  const int EW = cdiv(10 * HWSZ, 256);

  // ================= forward =================
  { const float* s2[2] = { xh0, xh1 }; composition(c, s2, 2, comphW, comphBN, comp_h); }
  { const float* s6[6] = { xp_[0], xp_[1], xp_[2], xp_[3], xp_[4], xp_[5] };
    composition(c, s6, 6, comppW, comppBN, comp_p); }
  { const float* m2[2] = { comp_h, comp_p }; conv_update(c, xf_fea, xf, m2, 2, updF, xf_n_o); }

  { const float* s4[4] = { xp_[0], xp_[1], xp_[2], xp_[3] };
    composition(c, s4, 4, compuW, compuBN, comp_u); }
  part_dep(c, xh1, xh0, dpU, dpu);
  decomposition(c, xf, xh0, dfhuW, dfhuB, att_o);                     // att_u
  k_bcast_mul<<<EW, 256, 0, stream>>>(att_o, xf, axf_u, 10 * HWSZ);
  { const float* m3[3] = { comp_u, axf_u, dpu }; conv_update(c, xh_fea, xh0, m3, 3, updU, xh_u_o); }

  { const float* s2[2] = { xp_[4], xp_[5] }; composition(c, s2, 2, complW, complBN, comp_l); }
  part_dep(c, xh0, xh1, dpL, dpl);
  decomposition(c, xf, xh1, dfhlW, dfhlB, att_o + HWSZ);              // att_l
  k_bcast_mul<<<EW, 256, 0, stream>>>(att_o + HWSZ, xf, axf_l, 10 * HWSZ);
  { const float* m3[3] = { comp_l, axf_l, dpl }; conv_update(c, xh_fea, xh1, m3, 3, updU, xh_l_o); }

  for (int i = 0; i < 6; ++i) {
    decomposition(c, xf, xp_[i], dfpW[i], dfpB[i], attfp[i]);
    k_bcast_mul<<<EW, 256, 0, stream>>>(attfp[i], xf, decfp[i], 10 * HWSZ);
    const float* src = (i < 4) ? xh0 : xh1;
    decomposition(c, src, xp_[i], dhpW[i], dhpB[i], atthp[i]);
    k_bcast_mul<<<EW, 256, 0, stream>>>(atthp[i], src, dechp[i], 10 * HWSZ);
  }

  static const int ES[10][2] = { {0,1},{1,0},{1,2},{1,3},{1,4},{1,5},
                                 {2,1},{3,1},{4,1},{5,1} };
  bool have[6] = { false, false, false, false, false, false };
  for (int e = 0; e < 10; ++e) {
    int s = ES[e][0], d = ES[e][1];
    if (!have[d]) { part_dep(c, xp_[s], xp_[d], dpE[e], msgs[d]); have[d] = true; }
    else {
      part_dep(c, xp_[s], xp_[d], dpE[e], c.MSGT);
      k_add_ip<<<EW, 256, 0, stream>>>(msgs[d], c.MSGT, 10 * HWSZ);
    }
  }

  for (int j = 0; j < 6; ++j) {
    const float* m3[3] = { decfp[j], dechp[j], msgs[j] };
    conv_update(c, xp_fea, xp_[j], m3, 3, updP[j], out + j * 10 * HWSZ);
  }

  for (int i = 0; i < 6; ++i)
    k_avg2<<<cdiv(HWSZ, 256), 256, 0, stream>>>(attfp[i], atthp[i],
                                                att_o + (2 + i) * HWSZ, HWSZ);
}
